// GraphTransformerLayer_79499844649021
// MI455X (gfx1250) — compile-verified, matrix operand order checked by reading, not run on hardware
//
#include <hip/hip_runtime.h>
#include <math.h>

// ---------------------------------------------------------------------------
// Types for CDNA5 WMMA
// ---------------------------------------------------------------------------
typedef __bf16 bf16;
typedef bf16  v16bf __attribute__((ext_vector_type(16)));
typedef float v8f   __attribute__((ext_vector_type(8)));

#define P_  4
#define C_  128
#define HH  8
#define DD  16

__device__ __forceinline__ bf16 f2bf(float x) {
    union { unsigned u; float f; } a; a.f = x;
    unsigned r = (a.u + 0x7FFFu + ((a.u >> 16) & 1u)) >> 16;   // RNE
    union { unsigned short s; bf16 b; } o; o.s = (unsigned short)r;
    return o.b;
}

__device__ __forceinline__ float redsum16(float v) {
    v += __shfl_xor(v, 1);
    v += __shfl_xor(v, 2);
    v += __shfl_xor(v, 4);
    v += __shfl_xor(v, 8);
    return v;
}

// float atomic-max via monotonic int/uint ordering trick (init must be -inf)
__device__ __forceinline__ void atomicMaxF(float* addr, float v) {
    if (v >= 0.0f) atomicMax((int*)addr, __float_as_int(v));
    else           atomicMin((unsigned int*)addr, __float_as_uint(v));
}

// ---------------------------------------------------------------------------
// Weight pre-swizzle: row-major fp32 (K x N) -> bf16 WMMA B-fragment order.
// Fragment flat index: (((kt*ntiles)+nt)*32 + lane)*16 + j
//   lane 0-15 : n = nt*16 + lane,      k = kt*32 + {0..7, 16..23}[j]
//   lane16-31 : n = nt*16 + lane-16,   k = kt*32 + {8..15,24..31}[j]
// ---------------------------------------------------------------------------
__global__ void prep_weight_kernel(const float* __restrict__ W, bf16* __restrict__ Bf,
                                   int K, int N) {
    int idx = blockIdx.x * 256 + threadIdx.x;
    if (idx >= K * N) return;
    int j    = idx & 15;
    int lane = (idx >> 4) & 31;
    int t    = idx >> 9;
    int ntiles = N >> 4;
    int nt = t % ntiles;
    int kt = t / ntiles;
    int hlf = lane >> 4;
    int n = nt * 16 + (lane & 15);
    int k = kt * 32 + hlf * 8 + (j < 8 ? j : j + 8);
    Bf[idx] = f2bf(W[(size_t)k * N + n]);
}

// ---------------------------------------------------------------------------
// Fused multi-output bf16 WMMA GEMM for K=128, N=128, M % 16 == 0:
//   O_i[M,128] = A[M,128] @ B_i      for i < NOUT (compile-time: 2 or 4)
// One LDS staging of the A tile feeds up to 4 weight matrices -> 4x fewer
// HBM reads of A. NOUT is a template parameter so the NOUT WMMAs per K-step
// issue back-to-back with no branches and no hazard NOPs (independent accs).
// block = 256 threads = 8 waves; wave w owns n-tile w.
// ---------------------------------------------------------------------------
template <int NOUT>
__global__ __launch_bounds__(256)
void gemm_wmma_bf16_multi(const float* __restrict__ A,
                          const bf16* __restrict__ B0, const bf16* __restrict__ B1,
                          const bf16* __restrict__ B2, const bf16* __restrict__ B3,
                          float* __restrict__ O0, float* __restrict__ O1,
                          float* __restrict__ O2, float* __restrict__ O3) {
    __shared__ bf16 As[16 * 128];

    const size_t row0 = (size_t)blockIdx.x * 16;
    const int tid = threadIdx.x;

    // cooperative stage: 16 x 128 fp32 -> bf16 in LDS (no guards: M % 16 == 0)
    for (int idx = tid * 4; idx < 16 * 128; idx += 256 * 4) {
        int r = idx >> 7, c = idx & 127;
        float4 v = *(const float4*)(A + (row0 + r) * C_ + c);
        As[idx + 0] = f2bf(v.x); As[idx + 1] = f2bf(v.y);
        As[idx + 2] = f2bf(v.z); As[idx + 3] = f2bf(v.w);
    }
    __syncthreads();

    const int wave = tid >> 5;
    const int lane = tid & 31;
    const int hlf  = lane >> 4;
    const int r    = lane & 15;
    const int nt   = wave;                 // 8 n-tiles of N=128

    const v16bf* Bv0 = (const v16bf*)B0;
    const v16bf* Bv1 = (const v16bf*)B1;
    const v16bf* Bv2 = (const v16bf*)B2;
    const v16bf* Bv3 = (const v16bf*)B3;

    v8f a0 = {0.f,0.f,0.f,0.f,0.f,0.f,0.f,0.f};
    v8f a1 = a0, a2 = a0, a3 = a0;

    #pragma unroll
    for (int kt = 0; kt < 4; ++kt) {
        v16bf a;
        const int base = r * 128 + kt * 32 + hlf * 8;
        #pragma unroll
        for (int j = 0; j < 8; ++j) a[j] = As[base + j];
        #pragma unroll
        for (int j = 0; j < 8; ++j) a[8 + j] = As[base + 16 + j];

        const size_t fi = ((size_t)kt * 8 + nt) * 32 + lane;
        a0 = __builtin_amdgcn_wmma_f32_16x16x32_bf16(false, a, false, Bv0[fi],
                                                     (short)0, a0, false, false);
        if (NOUT > 1)
            a1 = __builtin_amdgcn_wmma_f32_16x16x32_bf16(false, a, false, Bv1[fi],
                                                         (short)0, a1, false, false);
        if (NOUT > 2)
            a2 = __builtin_amdgcn_wmma_f32_16x16x32_bf16(false, a, false, Bv2[fi],
                                                         (short)0, a2, false, false);
        if (NOUT > 3)
            a3 = __builtin_amdgcn_wmma_f32_16x16x32_bf16(false, a, false, Bv3[fi],
                                                         (short)0, a3, false, false);
    }

    const int col = nt * 16 + r;
    #pragma unroll
    for (int i = 0; i < 8; ++i) {
        size_t o = (row0 + hlf * 8 + i) * C_ + col;
        O0[o] = a0[i];
        if (NOUT > 1) O1[o] = a1[i];
        if (NOUT > 2) O2[o] = a2[i];
        if (NOUT > 3) O3[o] = a3[i];
    }
}

// ---------------------------------------------------------------------------
// Generic bf16 WMMA GEMM:  C[M,N] = act(A[M,K] @ B + bias) + resid
// (used for the projections with bias/residual/relu epilogues and K/N != 128)
// ---------------------------------------------------------------------------
__global__ __launch_bounds__(256)
void gemm_wmma_bf16(const float* __restrict__ A, const bf16* __restrict__ Bf,
                    const float* __restrict__ bias, const float* __restrict__ resid,
                    float* __restrict__ Cout, int M, int K, int N, int doRelu) {
    __shared__ bf16 As[16 * 256];          // up to K=256

    const int row0 = blockIdx.x * 16;
    const int tid  = threadIdx.x;
    const int total = 16 * K;

    for (int idx = tid * 4; idx < total; idx += 256 * 4) {
        int r = idx / K, c = idx % K;
        int grow = row0 + r;
        float4 v = make_float4(0.f, 0.f, 0.f, 0.f);
        if (grow < M) v = *(const float4*)(A + (size_t)grow * K + c);
        As[idx + 0] = f2bf(v.x); As[idx + 1] = f2bf(v.y);
        As[idx + 2] = f2bf(v.z); As[idx + 3] = f2bf(v.w);
    }
    __syncthreads();

    const int wave = tid >> 5;
    const int lane = tid & 31;
    const int hlf  = lane >> 4;
    const int r    = lane & 15;
    const int ntiles = N >> 4;
    const int nt = blockIdx.y * 8 + wave;
    if (nt >= ntiles) return;

    const v16bf* Bv = (const v16bf*)Bf;
    const int kts = K >> 5;

    v8f acc = {0.f, 0.f, 0.f, 0.f, 0.f, 0.f, 0.f, 0.f};
    for (int kt = 0; kt < kts; ++kt) {
        if (kt + 1 < kts)
            __builtin_prefetch(Bv + ((size_t)(kt + 1) * ntiles + nt) * 32 + lane, 0, 1);
        v16bf a;
        const int base = r * K + kt * 32 + hlf * 8;
        #pragma unroll
        for (int j = 0; j < 8; ++j) a[j] = As[base + j];
        #pragma unroll
        for (int j = 0; j < 8; ++j) a[8 + j] = As[base + 16 + j];
        v16bf b = Bv[((size_t)kt * ntiles + nt) * 32 + lane];
        acc = __builtin_amdgcn_wmma_f32_16x16x32_bf16(
                  false, a, false, b, (short)0, acc, false, false);
    }

    const int col = nt * 16 + r;
    const float bcol = bias ? bias[col] : 0.0f;
    #pragma unroll
    for (int i = 0; i < 8; ++i) {
        int grow = row0 + hlf * 8 + i;
        if (grow < M) {
            float v = acc[i] + bcol;
            if (doRelu) v = fmaxf(v, 0.0f);
            if (resid)  v += resid[(size_t)grow * N + col];
            Cout[(size_t)grow * N + col] = v;
        }
    }
}

// ---------------------------------------------------------------------------
// misc elementwise / graph kernels
// ---------------------------------------------------------------------------
__global__ void fill_kernel(float* p, float v, size_t n) {
    size_t i = (size_t)blockIdx.x * blockDim.x + threadIdx.x;
    if (i < n) p[i] = v;
}

// one thread per (edge, p, c): score/e_out, softmax weights, wV/z scatter, diff
__global__ void edge_attn_kernel(const float* __restrict__ h,
                                 const float* __restrict__ Q,
                                 const float* __restrict__ Km,
                                 const float* __restrict__ V,
                                 float* __restrict__ score,   // in: proj_e, out: score
                                 const int* __restrict__ src, const int* __restrict__ dst,
                                 float* __restrict__ wV, float* __restrict__ z,
                                 float* __restrict__ diff, int nEdges) {
    size_t gid = (size_t)blockIdx.x * 256 + threadIdx.x;
    size_t total = (size_t)nEdges * P_ * C_;
    if (gid >= total) return;
    int c = gid & (C_ - 1);
    size_t ep = gid >> 7;          // e*P + p
    int e = (int)(ep >> 2), p = (int)(ep & 3);
    int s_ = src[e], d_ = dst[e];
    size_t si = ((size_t)s_ * P_ + p) * C_ + c;
    size_t di = ((size_t)d_ * P_ + p) * C_ + c;

    float sc = Km[si] * Q[di] * 0.25f + score[gid];   // 1/sqrt(D)=0.25
    score[gid] = sc;                                  // e_out

    float ssum = redsum16(sc);                        // sum over D within head
    float sexp = expf(fminf(fmaxf(ssum, -5.0f), 5.0f));

    atomicAdd(&wV[di], V[si] * sexp);
    if ((c & (DD - 1)) == 0)
        atomicAdd(&z[((size_t)d_ * P_ + p) * HH + (c >> 4)], sexp);

    diff[gid] = fmaxf(h[si] - h[di], 0.0f);
}

// one thread per (edge, p, c): cosine-scaled message + segment-max scatter
__global__ void edge_enhance_kernel(const float* __restrict__ Qe,
                                    const float* __restrict__ kd,
                                    const float* __restrict__ vd,
                                    const float* __restrict__ repe,
                                    const int* __restrict__ dst,
                                    float* __restrict__ h_enh, int nEdges) {
    size_t gid = (size_t)blockIdx.x * 256 + threadIdx.x;
    size_t total = (size_t)nEdges * P_ * C_;
    if (gid >= total) return;
    int c = gid & (C_ - 1);
    size_t ep = gid >> 7;
    int e = (int)(ep >> 2), p = (int)(ep & 3);
    int d_ = dst[e];
    size_t di = ((size_t)d_ * P_ + p) * C_ + c;

    float q = Qe[di];
    float k = kd[gid];
    float nk = sqrtf(redsum16(k * k) + 1e-12f);
    float nq = sqrtf(redsum16(q * q) + 1e-12f);
    float sc = q * k * (nk * nq + 1e-6f) + repe[gid];
    float msg = sc * vd[gid];
    atomicMaxF(&h_enh[di], msg);
}

// float4-vectorized combine: one thread per 4 channels
__global__ void combine_kernel(const float* __restrict__ wV, const float* __restrict__ z,
                               const float* __restrict__ h_enh,
                               float* __restrict__ hf_pre, size_t total4) {
    size_t gid = (size_t)blockIdx.x * 256 + threadIdx.x;
    if (gid >= total4) return;
    int c4 = (int)(gid & (C_ / 4 - 1)) * 4;      // 4 consecutive channels, same head
    size_t np = gid >> 5;
    float zz = z[np * HH + (c4 >> 4)] + 1e-6f;
    float4 w  = ((const float4*)wV)[gid];
    float4 en = ((const float4*)h_enh)[gid];
    float o[4] = { w.x / zz, w.y / zz, w.z / zz, w.w / zz };
    float ee[4] = { en.x, en.y, en.z, en.w };
    #pragma unroll
    for (int i = 0; i < 4; ++i) {
        float v = ee[i];
        if (!(v > -1e37f) || !(v < 1e37f)) v = 0.0f;   // -inf init / NaN -> 0
        o[i] += v;
    }
    ((float4*)hf_pre)[gid] = make_float4(o[0], o[1], o[2], o[3]);
}

// two-pass batch-norm: column sums into stats[0..127]=sum, [128..255]=sumsq
__global__ void bn_reduce_kernel(const float* __restrict__ X, float* __restrict__ stats,
                                 size_t M) {
    __shared__ float ssum[C_], ssq[C_];
    int tid = threadIdx.x;
    if (tid < C_) { ssum[tid] = 0.f; ssq[tid] = 0.f; }
    __syncthreads();
    int c = tid & (C_ - 1);
    float ls = 0.f, lq = 0.f;
    for (size_t row = (size_t)blockIdx.x * 2 + (tid >> 7); row < M;
         row += (size_t)gridDim.x * 2) {
        float v = X[row * C_ + c];
        ls += v; lq += v * v;
    }
    atomicAdd(&ssum[c], ls); atomicAdd(&ssq[c], lq);
    __syncthreads();
    if (tid < C_) {
        atomicAdd(&stats[tid],       ssum[tid]);
        atomicAdd(&stats[C_ + tid],  ssq[tid]);
    }
}

// float4-vectorized BN apply
__global__ void bn_apply_kernel(const float* __restrict__ X, float* __restrict__ Y,
                                const float* __restrict__ stats,
                                const float* __restrict__ g, const float* __restrict__ b,
                                size_t M) {
    size_t gid = (size_t)blockIdx.x * 256 + threadIdx.x;
    size_t total4 = M * (C_ / 4);
    if (gid >= total4) return;
    int c4 = (int)(gid & (C_ / 4 - 1)) * 4;
    float inv = 1.0f / (float)M;
    float4 x = ((const float4*)X)[gid];
    float xs[4] = { x.x, x.y, x.z, x.w };
    float o[4];
    #pragma unroll
    for (int i = 0; i < 4; ++i) {
        int c = c4 + i;
        float mean = stats[c] * inv;
        float var  = stats[C_ + c] * inv - mean * mean;
        o[i] = (xs[i] - mean) * rsqrtf(var + 1e-5f) * g[c] + b[c];
    }
    ((float4*)Y)[gid] = make_float4(o[0], o[1], o[2], o[3]);
}

// ---------------------------------------------------------------------------
// Host-side orchestration
// ---------------------------------------------------------------------------
extern "C" void kernel_launch(void* const* d_in, const int* in_sizes, int n_in,
                              void* d_out, int out_size, void* d_ws, size_t ws_size,
                              hipStream_t stream) {
    const float* h    = (const float*)d_in[0];
    const float* e    = (const float*)d_in[1];
    const int*   src  = (const int*)d_in[2];
    const int*   dst  = (const int*)d_in[3];
    const float* Wq   = (const float*)d_in[4];
    const float* Wk   = (const float*)d_in[5];
    const float* Wv   = (const float*)d_in[6];
    const float* We   = (const float*)d_in[7];
    const float* eWq  = (const float*)d_in[8];
    const float* eWk  = (const float*)d_in[9];
    const float* eWv  = (const float*)d_in[10];
    const float* eWre = (const float*)d_in[11];
    const float* Oh_w = (const float*)d_in[12];
    const float* Oh_b = (const float*)d_in[13];
    const float* Oe_w = (const float*)d_in[14];
    const float* Oe_b = (const float*)d_in[15];
    const float* bn_h_g = (const float*)d_in[16];
    const float* bn_h_b = (const float*)d_in[17];
    const float* bn_e_g = (const float*)d_in[18];
    const float* bn_e_b = (const float*)d_in[19];
    const float* W1   = (const float*)d_in[20];
    const float* b1   = (const float*)d_in[21];
    const float* W2   = (const float*)d_in[22];
    const float* b2   = (const float*)d_in[23];

    const int nNodes = in_sizes[0] / (P_ * C_);
    const int nEdges = in_sizes[2];
    const size_t nodeRows = (size_t)nNodes * P_;     // 40000
    const size_t edgeRows = (size_t)nEdges * P_;     // 320000

    float* out_h = (float*)d_out;                    // (nodeRows, C)
    float* out_e = out_h + nodeRows * C_;            // (edgeRows, C)

    // ---- workspace layout ----
    char* wsb = (char*)d_ws;
    bf16* frag = (bf16*)wsb;
    const size_t FRAG_SZ[12] = {16384,16384,16384,16384,16384,16384,
                                16384,16384,16384,16384,32768,32768};
    size_t fragOff[12]; size_t fo = 0;
    for (int i = 0; i < 12; ++i) { fragOff[i] = fo; fo += FRAG_SZ[i]; }
    float* fp = (float*)(wsb + ((fo * sizeof(bf16) + 255) & ~(size_t)255));
    auto alloc = [&](size_t n) { float* p = fp; fp += (n + 63) & ~(size_t)63; return p; };

    float* Qb     = alloc(nodeRows * C_);
    float* Kb     = alloc(nodeRows * C_);
    float* Vb     = alloc(nodeRows * C_);
    float* Qeb    = alloc(nodeRows * C_);
    float* wV     = alloc(nodeRows * C_);
    float* zb     = alloc(nodeRows * HH);
    float* h_enh  = alloc(nodeRows * C_);
    float* hf_pre = alloc(nodeRows * C_);
    float* h2     = alloc(nodeRows * C_);
    float* ffnmid = alloc(nodeRows * 2 * C_);
    float* scoreb = alloc(edgeRows * C_);
    float* repeb  = alloc(edgeRows * C_);
    float* diffb  = alloc(edgeRows * C_);
    float* kdiff  = alloc(edgeRows * C_);
    float* vdiff  = alloc(edgeRows * C_);
    float* statsH = alloc(256);
    float* statsE = alloc(256);

    // ---- init accumulators ----
    hipMemsetAsync(wV,     0, nodeRows * C_ * sizeof(float), stream);
    hipMemsetAsync(zb,     0, nodeRows * HH * sizeof(float), stream);
    hipMemsetAsync(statsH, 0, 256 * sizeof(float), stream);
    hipMemsetAsync(statsE, 0, 256 * sizeof(float), stream);
    {
        size_t n = nodeRows * C_;
        fill_kernel<<<(unsigned)((n + 255) / 256), 256, 0, stream>>>(
            h_enh, -__builtin_huge_valf(), n);
    }

    // ---- pre-swizzle weights to bf16 WMMA fragments ----
    auto prep = [&](const float* W, int slot, int K, int N) {
        int tot = K * N;
        prep_weight_kernel<<<(tot + 255) / 256, 256, 0, stream>>>(
            W, frag + fragOff[slot], K, N);
    };
    prep(Wq, 0, 128, 128);  prep(Wk, 1, 128, 128);  prep(Wv, 2, 128, 128);
    prep(We, 3, 128, 128);  prep(eWq, 4, 128, 128); prep(eWk, 5, 128, 128);
    prep(eWv, 6, 128, 128); prep(eWre, 7, 128, 128);
    prep(Oh_w, 8, 128, 128); prep(Oe_w, 9, 128, 128);
    prep(W1, 10, 128, 256);  prep(W2, 11, 256, 128);

    auto gemm = [&](const float* A, int slot, const float* bias, const float* resid,
                    float* Cc, size_t M, int K, int N, int relu) {
        dim3 g((unsigned)(M / 16), (unsigned)(((N / 16) + 7) / 8));
        gemm_wmma_bf16<<<g, 256, 0, stream>>>(A, frag + fragOff[slot], bias, resid,
                                              Cc, (int)M, K, N, relu);
    };

    // ---- fused node projections: Q, K, V, Qe (single pass over h) ----
    gemm_wmma_bf16_multi<4><<<(unsigned)(nodeRows / 16), 256, 0, stream>>>(
        h, frag + fragOff[0], frag + fragOff[1], frag + fragOff[2], frag + fragOff[4],
        Qb, Kb, Vb, Qeb);

    // ---- fused edge projections: proj_e (-> score buf), repe (single pass over e) ----
    gemm_wmma_bf16_multi<2><<<(unsigned)(edgeRows / 16), 256, 0, stream>>>(
        e, frag + fragOff[3], frag + fragOff[7], frag + fragOff[3], frag + fragOff[7],
        scoreb, repeb, nullptr, nullptr);

    // ---- attention gather/scatter + diff ----
    {
        size_t total = edgeRows * C_;
        edge_attn_kernel<<<(unsigned)((total + 255) / 256), 256, 0, stream>>>(
            h, Qb, Kb, Vb, scoreb, src, dst, wV, zb, diffb, nEdges);
    }

    // ---- fused enhance projections on diff (single pass over diff) ----
    gemm_wmma_bf16_multi<2><<<(unsigned)(edgeRows / 16), 256, 0, stream>>>(
        diffb, frag + fragOff[5], frag + fragOff[6], frag + fragOff[5], frag + fragOff[6],
        kdiff, vdiff, nullptr, nullptr);

    // ---- enhance scatter (segment max) ----
    {
        size_t total = edgeRows * C_;
        edge_enhance_kernel<<<(unsigned)((total + 255) / 256), 256, 0, stream>>>(
            Qeb, kdiff, vdiff, repeb, dst, h_enh, nEdges);
    }

    // ---- combine attn + enhance ----
    {
        size_t total4 = nodeRows * (C_ / 4);
        combine_kernel<<<(unsigned)((total4 + 255) / 256), 256, 0, stream>>>(
            wV, zb, h_enh, hf_pre, total4);
    }

    // ---- node output projection + residual, then BN -> h2 ----
    gemm(hf_pre, 8, Oh_b, h, out_h, nodeRows, 128, 128, 0);
    bn_reduce_kernel<<<512, 256, 0, stream>>>(out_h, statsH, nodeRows);
    {
        size_t total4 = nodeRows * (C_ / 4);
        bn_apply_kernel<<<(unsigned)((total4 + 255) / 256), 256, 0, stream>>>(
            out_h, h2, statsH, bn_h_g, bn_h_b, nodeRows);
    }

    // ---- FFN: relu(h2@W1+b1)@W2 + b2 + h2 -> out_h ----
    gemm(h2, 10, b1, nullptr, ffnmid, nodeRows, 128, 256, 1);
    gemm(ffnmid, 11, b2, h2, out_h, nodeRows, 256, 128, 0);

    // ---- edge output projection + residual, then BN in place ----
    gemm(scoreb, 9, Oe_b, e, out_e, edgeRows, 128, 128, 0);
    bn_reduce_kernel<<<1024, 256, 0, stream>>>(out_e, statsE, edgeRows);
    {
        size_t total4 = edgeRows * (C_ / 4);
        bn_apply_kernel<<<(unsigned)((total4 + 255) / 256), 256, 0, stream>>>(
            out_e, out_e, statsE, bn_e_g, bn_e_b, edgeRows);
    }
}
